// HierarchicalHeteroGraphSage_8237747274315
// MI455X (gfx1250) — matrix operand order repaired and checked by reading, not compiled, not run on hardware
//
#include <hip/hip_runtime.h>

#define C0N 50000
#define C1N 150000
#define C2N 300000
#define E0N 300000
#define E1N 600000
#define EALL (E0N + E1N)
#define CINC 64
#define HIDC 128
#define LN_EPS 1e-5f
#define L2_EPS 1e-12f

typedef __attribute__((ext_vector_type(16))) _Float16 v16h;
typedef __attribute__((ext_vector_type(8)))  float    v8f;

// ---------------------------------------------------------------------------
// WMMA helper: D = A(16x32 f16) * B(32x16 f16) + C(16x16 f32)
// ---------------------------------------------------------------------------
__device__ __forceinline__ v8f wmma16(v16h a, v16h b, v8f c) {
  return __builtin_amdgcn_wmma_f32_16x16x32_f16(
      /*neg_a=*/false, a, /*neg_b=*/false, b,
      /*c_mod=*/(short)0, c, /*reuse_a=*/false, /*reuse_b=*/false);
}

// A-fragment from f32 global memory (scaled, converted to f16).
// Contiguous 16-bit A layout: lanes 0-15 hold K=0..15; lanes 16-31 hold
// K=16..31.  p = rowBase + ktile*32 + half*16.
__device__ __forceinline__ v16h load_a16(const float* __restrict__ p, float s) {
  float4 f0 = *(const float4*)(p + 0);
  float4 f1 = *(const float4*)(p + 4);
  float4 f2 = *(const float4*)(p + 8);
  float4 f3 = *(const float4*)(p + 12);
  v16h r;
  r[0]  = (_Float16)(f0.x * s); r[1]  = (_Float16)(f0.y * s);
  r[2]  = (_Float16)(f0.z * s); r[3]  = (_Float16)(f0.w * s);
  r[4]  = (_Float16)(f1.x * s); r[5]  = (_Float16)(f1.y * s);
  r[6]  = (_Float16)(f1.z * s); r[7]  = (_Float16)(f1.w * s);
  r[8]  = (_Float16)(f2.x * s); r[9]  = (_Float16)(f2.y * s);
  r[10] = (_Float16)(f2.z * s); r[11] = (_Float16)(f2.w * s);
  r[12] = (_Float16)(f3.x * s); r[13] = (_Float16)(f3.y * s);
  r[14] = (_Float16)(f3.z * s); r[15] = (_Float16)(f3.w * s);
  return r;
}

// Stage W (K x N f32 row-major) into LDS transposed + f16: Wt[n*K + k].
template <int K, int N>
__device__ __forceinline__ void stage_weights(const float* __restrict__ W,
                                              _Float16* Wt, int tid) {
#pragma unroll 4
  for (int idx = tid; idx < K * N; idx += 256) {
    int k = idx / N;            // N is a power of two -> shift
    int n = idx & (N - 1);
    Wt[n * K + k] = (_Float16)W[idx];
  }
}

// ---------------------------------------------------------------------------
// Kernel 1: xp = relu(x @ Wp + bp)     M x 64  @  64 x 64
// ---------------------------------------------------------------------------
template <bool G>
__device__ __forceinline__ void xp_body(const float* __restrict__ x,
                                        const _Float16* Wt, const float* bs,
                                        float* __restrict__ out, int M,
                                        int base, int lane) {
  int kh = lane >> 4, e = lane & 15;
  int aRow = base + e;
  if (G) aRow = min(aRow, M - 1);
  const float* ap = x + (size_t)aRow * CINC + kh * 16;
  v16h a0 = load_a16(ap, 1.0f);
  v16h a1 = load_a16(ap + 32, 1.0f);

  float* op = out + (size_t)(base + kh * 8) * CINC + e;
#pragma unroll
  for (int nt = 0; nt < CINC / 16; ++nt) {
    v8f acc = {};
    const _Float16* wn = Wt + (nt * 16 + e) * CINC + kh * 16;
    acc = wmma16(a0, *(const v16h*)(wn + 0), acc);
    acc = wmma16(a1, *(const v16h*)(wn + 32), acc);
    float b = bs[nt * 16 + e];
#pragma unroll
    for (int r = 0; r < 8; ++r) {
      float v = fmaxf(acc[r] + b, 0.f);
      if (!G || (base + kh * 8 + r) < M)
        op[(size_t)(r * CINC + nt * 16)] = v;
    }
  }
}

__global__ __launch_bounds__(256) void k_xp(const float* __restrict__ x,
                                            const float* __restrict__ W,
                                            const float* __restrict__ bias,
                                            float* __restrict__ out, int M) {
  __shared__ __align__(32) _Float16 Wt[CINC * CINC];
  __shared__ float bs[CINC];
  int tid = threadIdx.x;
  stage_weights<CINC, CINC>(W, Wt, tid);
  if (tid < CINC) bs[tid] = bias[tid];
  __syncthreads();
  int base = blockIdx.x * 128 + (tid >> 5) * 16;
  int lane = tid & 31;
  if ((blockIdx.x + 1) * 128 <= M)
    xp_body<false>(x, Wt, bs, out, M, base, lane);
  else
    xp_body<true>(x, Wt, bs, out, M, base, lane);
}

// ---------------------------------------------------------------------------
// Kernel 2: o = L2norm( (aggsum/cnt) @ Wl + x @ Wr + bl );  hsum (+)= o
// ---------------------------------------------------------------------------
template <bool G>
__device__ __forceinline__ void o_body(const float* __restrict__ aggsum,
                                       const float* __restrict__ cnt,
                                       const float* __restrict__ x,
                                       const _Float16* Wt1, const _Float16* Wt2,
                                       const float* bs, float* __restrict__ hsum,
                                       int M, int accum, int base, int lane) {
  int kh = lane >> 4, e = lane & 15;
  int aRow = base + e;
  if (G) aRow = min(aRow, M - 1);
  float s = 1.0f / fmaxf(cnt[aRow], 1.0f);   // fused segment-mean divide
  const float* ap1 = aggsum + (size_t)aRow * CINC + kh * 16;
  const float* ap2 = x      + (size_t)aRow * CINC + kh * 16;
  v16h a10 = load_a16(ap1, s);
  v16h a11 = load_a16(ap1 + 32, s);
  v16h a20 = load_a16(ap2, 1.0f);
  v16h a21 = load_a16(ap2 + 32, 1.0f);

  v8f acc[HIDC / 16];
#pragma unroll
  for (int nt = 0; nt < HIDC / 16; ++nt) {
    v8f c = {};
    const _Float16* w1 = Wt1 + (nt * 16 + e) * CINC + kh * 16;
    const _Float16* w2 = Wt2 + (nt * 16 + e) * CINC + kh * 16;
    c = wmma16(a10, *(const v16h*)(w1 + 0), c);
    c = wmma16(a11, *(const v16h*)(w1 + 32), c);
    c = wmma16(a20, *(const v16h*)(w2 + 0), c);
    c = wmma16(a21, *(const v16h*)(w2 + 32), c);
    float b = bs[nt * 16 + e];
#pragma unroll
    for (int r = 0; r < 8; ++r) c[r] += b;
    acc[nt] = c;
  }

  // per-row L2 norm: register r of half kh is row base + r + kh*8, spread
  // over lanes e (cols) and nt tiles.
#pragma unroll
  for (int r = 0; r < 8; ++r) {
    float ss = 0.f;
#pragma unroll
    for (int nt = 0; nt < HIDC / 16; ++nt) ss += acc[nt][r] * acc[nt][r];
    for (int m = 1; m < 16; m <<= 1) ss += __shfl_xor(ss, m, 32);
    float inv = 1.0f / fmaxf(sqrtf(ss), L2_EPS);
#pragma unroll
    for (int nt = 0; nt < HIDC / 16; ++nt) acc[nt][r] *= inv;
  }

  float* op = hsum + (size_t)(base + kh * 8) * HIDC + e;
#pragma unroll
  for (int nt = 0; nt < HIDC / 16; ++nt) {
#pragma unroll
    for (int r = 0; r < 8; ++r) {
      if (!G || (base + kh * 8 + r) < M) {
        size_t off = (size_t)(r * HIDC + nt * 16);
        float v = acc[nt][r];
        op[off] = accum ? op[off] + v : v;
      }
    }
  }
}

__global__ __launch_bounds__(256) void k_o(const float* __restrict__ aggsum,
                                           const float* __restrict__ cnt,
                                           const float* __restrict__ x,
                                           const float* __restrict__ Wl,
                                           const float* __restrict__ Wr,
                                           const float* __restrict__ bias,
                                           float* __restrict__ hsum,
                                           int M, int accum) {
  __shared__ __align__(32) _Float16 Wt1[HIDC * CINC];
  __shared__ __align__(32) _Float16 Wt2[HIDC * CINC];
  __shared__ float bs[HIDC];
  int tid = threadIdx.x;
  stage_weights<CINC, HIDC>(Wl, Wt1, tid);
  stage_weights<CINC, HIDC>(Wr, Wt2, tid);
  if (tid < HIDC) bs[tid] = bias[tid];
  __syncthreads();
  int base = blockIdx.x * 128 + (tid >> 5) * 16;
  int lane = tid & 31;
  if ((blockIdx.x + 1) * 128 <= M)
    o_body<false>(aggsum, cnt, x, Wt1, Wt2, bs, hsum, M, accum, base, lane);
  else
    o_body<true>(aggsum, cnt, x, Wt1, Wt2, bs, hsum, M, accum, base, lane);
}

// ---------------------------------------------------------------------------
// Kernel 3: out (+)= (agg2/cnt2) @ Wl2 + h @ Wr2 + bl2   (M x 128, K = 128)
// ---------------------------------------------------------------------------
template <bool G>
__device__ __forceinline__ void h2_body(const float* __restrict__ agg2,
                                        const float* __restrict__ cnt2,
                                        const float* __restrict__ h,
                                        const _Float16* Wt1, const _Float16* Wt2,
                                        const float* bs, float* __restrict__ out,
                                        int M, int accum, int base, int lane) {
  int kh = lane >> 4, e = lane & 15;
  int aRow = base + e;
  if (G) aRow = min(aRow, M - 1);
  float s = 1.0f / fmaxf(cnt2[aRow], 1.0f);
  const float* ap1 = agg2 + (size_t)aRow * HIDC + kh * 16;
  const float* ap2 = h    + (size_t)aRow * HIDC + kh * 16;
  v16h a1[4], a2[4];
#pragma unroll
  for (int kt = 0; kt < 4; ++kt) {
    a1[kt] = load_a16(ap1 + kt * 32, s);
    a2[kt] = load_a16(ap2 + kt * 32, 1.0f);
  }

  float* op = out + (size_t)(base + kh * 8) * HIDC + e;
#pragma unroll
  for (int nt = 0; nt < HIDC / 16; ++nt) {
    v8f c = {};
    const _Float16* w1 = Wt1 + (nt * 16 + e) * HIDC + kh * 16;
    const _Float16* w2 = Wt2 + (nt * 16 + e) * HIDC + kh * 16;
#pragma unroll
    for (int kt = 0; kt < 4; ++kt) c = wmma16(a1[kt], *(const v16h*)(w1 + kt * 32), c);
#pragma unroll
    for (int kt = 0; kt < 4; ++kt) c = wmma16(a2[kt], *(const v16h*)(w2 + kt * 32), c);
    float b = bs[nt * 16 + e];
#pragma unroll
    for (int r = 0; r < 8; ++r) {
      if (!G || (base + kh * 8 + r) < M) {
        size_t off = (size_t)(r * HIDC + nt * 16);
        float v = c[r] + b;
        op[off] = accum ? op[off] + v : v;
      }
    }
  }
}

__global__ __launch_bounds__(256) void k_h2(const float* __restrict__ agg2,
                                            const float* __restrict__ cnt2,
                                            const float* __restrict__ h,
                                            const float* __restrict__ Wl2,
                                            const float* __restrict__ Wr2,
                                            const float* __restrict__ bias,
                                            float* __restrict__ out,
                                            int M, int accum) {
  __shared__ __align__(32) _Float16 Wt1[HIDC * HIDC];
  __shared__ __align__(32) _Float16 Wt2[HIDC * HIDC];
  __shared__ float bs[HIDC];
  int tid = threadIdx.x;
  stage_weights<HIDC, HIDC>(Wl2, Wt1, tid);
  stage_weights<HIDC, HIDC>(Wr2, Wt2, tid);
  if (tid < HIDC) bs[tid] = bias[tid];
  __syncthreads();
  int base = blockIdx.x * 128 + (tid >> 5) * 16;
  int lane = tid & 31;
  if ((blockIdx.x + 1) * 128 <= M)
    h2_body<false>(agg2, cnt2, h, Wt1, Wt2, bs, out, M, accum, base, lane);
  else
    h2_body<true>(agg2, cnt2, h, Wt1, Wt2, bs, out, M, accum, base, lane);
}

// ---------------------------------------------------------------------------
// Scatter: sum[dst] += feat[src]  (float4 per thread), cnt[dst] += 1
// ---------------------------------------------------------------------------
__global__ void k_scatter(const float* __restrict__ feat,
                          const int* __restrict__ src,
                          const int* __restrict__ dst,
                          float* __restrict__ sum, float* __restrict__ cnt,
                          int E, int Cch, int lg4) {
  long long tid = (long long)blockIdx.x * blockDim.x + threadIdx.x;
  int c4 = Cch >> 2;
  long long e = tid >> lg4;
  int q = (int)(tid & (c4 - 1));
  if (e >= E) return;
  int sIdx = src[e], dIdx = dst[e];
  float4 v = *(const float4*)(feat + (size_t)sIdx * Cch + q * 4);
  float* p = sum + (size_t)dIdx * Cch + q * 4;
  atomicAdd(p + 0, v.x);
  atomicAdd(p + 1, v.y);
  atomicAdd(p + 2, v.z);
  atomicAdd(p + 3, v.w);
  if (q == 0) atomicAdd(&cnt[dIdx], 1.0f);
}

__global__ void k_zero(float* __restrict__ p, size_t n) {
  size_t i = (size_t)blockIdx.x * blockDim.x + threadIdx.x;
  if (i < n) p[i] = 0.f;
}

// ---------------------------------------------------------------------------
// relu + mean(/3) + LayerNorm over 128 channels, one wave per row, in place
// ---------------------------------------------------------------------------
__global__ __launch_bounds__(256) void k_relu_ln(float* __restrict__ buf,
                                                 const float* __restrict__ g,
                                                 const float* __restrict__ b,
                                                 int M, float scale) {
  int wave = threadIdx.x >> 5, lane = threadIdx.x & 31;
  int row = blockIdx.x * 8 + wave;
  if (row >= M) return;
  float* p = buf + (size_t)row * HIDC;
  float4 v = *(const float4*)(p + lane * 4);
  v.x = fmaxf(v.x * scale, 0.f); v.y = fmaxf(v.y * scale, 0.f);
  v.z = fmaxf(v.z * scale, 0.f); v.w = fmaxf(v.w * scale, 0.f);
  float sum = v.x + v.y + v.z + v.w;
  float sq  = v.x * v.x + v.y * v.y + v.z * v.z + v.w * v.w;
  for (int m = 1; m < 32; m <<= 1) {
    sum += __shfl_xor(sum, m, 32);
    sq  += __shfl_xor(sq,  m, 32);
  }
  float mu  = sum * (1.0f / HIDC);
  float var = sq * (1.0f / HIDC) - mu * mu;
  float rn  = rsqrtf(var + LN_EPS);
  float4 gg = *(const float4*)(g + lane * 4);
  float4 bb = *(const float4*)(b + lane * 4);
  v.x = (v.x - mu) * rn * gg.x + bb.x;
  v.y = (v.y - mu) * rn * gg.y + bb.y;
  v.z = (v.z - mu) * rn * gg.z + bb.z;
  v.w = (v.w - mu) * rn * gg.w + bb.w;
  *(float4*)(p + lane * 4) = v;
}

// ---------------------------------------------------------------------------
extern "C" void kernel_launch(void* const* d_in, const int* in_sizes, int n_in,
                              void* d_out, int out_size, void* d_ws, size_t ws_size,
                              hipStream_t stream) {
  const float* x   = (const float*)d_in[0];
  const int*   ei[3] = {(const int*)d_in[1], (const int*)d_in[2], (const int*)d_in[3]};
  const float* Wp  = (const float*)d_in[6];
  const float* bp  = (const float*)d_in[7];
  const float* Wl  = (const float*)d_in[8];
  const float* bl  = (const float*)d_in[9];
  const float* Wr  = (const float*)d_in[10];
  const float* Wl2 = (const float*)d_in[11];
  const float* bl2 = (const float*)d_in[12];
  const float* Wr2 = (const float*)d_in[13];
  const float* g0  = (const float*)d_in[14];
  const float* b0  = (const float*)d_in[15];
  const float* g1  = (const float*)d_in[16];
  const float* b1  = (const float*)d_in[17];
  float* out = (float*)d_out;

  // workspace layout (floats); aggsum/cnt1 and agg2/cnt2 kept adjacent so one
  // zero-fill covers both
  float* ws = (float*)d_ws;
  size_t o = 0;
  float* xp     = ws + o; o += (size_t)C2N * CINC;
  float* aggsum = ws + o; o += (size_t)C1N * CINC;
  float* cnt1   = ws + o; o += (size_t)C1N;
  float* hbuf   = ws + o; o += (size_t)C1N * HIDC;
  float* agg2   = ws + o; o += (size_t)C0N * HIDC;
  float* cnt2   = ws + o; o += (size_t)C0N;

  dim3 blk(256);

  for (int t = 0; t < 3; ++t) {
    k_xp<<<dim3((C2N + 127) / 128), blk, 0, stream>>>(
        x, Wp + (size_t)t * CINC * CINC, bp + (size_t)t * CINC, xp, C2N);

    size_t nz1 = (size_t)C1N * CINC + C1N;
    k_zero<<<dim3((unsigned)((nz1 + 255) / 256)), blk, 0, stream>>>(aggsum, nz1);

    long long nthr = (long long)EALL * (CINC / 4);
    k_scatter<<<dim3((unsigned)((nthr + 255) / 256)), blk, 0, stream>>>(
        xp, ei[t], ei[t] + EALL, aggsum, cnt1, EALL, CINC, 4);

    k_o<<<dim3((C1N + 127) / 128), blk, 0, stream>>>(
        aggsum, cnt1, x, Wl + (size_t)t * CINC * HIDC, Wr + (size_t)t * CINC * HIDC,
        bl + (size_t)t * HIDC, hbuf, C1N, t);
  }

  k_relu_ln<<<dim3((C1N + 7) / 8), blk, 0, stream>>>(hbuf, g0, b0, C1N, 1.0f / 3.0f);

  for (int t = 0; t < 3; ++t) {
    size_t nz2 = (size_t)C0N * HIDC + C0N;
    k_zero<<<dim3((unsigned)((nz2 + 255) / 256)), blk, 0, stream>>>(agg2, nz2);

    long long nthr = (long long)E0N * (HIDC / 4);
    k_scatter<<<dim3((unsigned)((nthr + 255) / 256)), blk, 0, stream>>>(
        hbuf, ei[t], ei[t] + EALL, agg2, cnt2, E0N, HIDC, 5);

    k_h2<<<dim3((C0N + 127) / 128), blk, 0, stream>>>(
        agg2, cnt2, hbuf, Wl2 + (size_t)t * HIDC * HIDC, Wr2 + (size_t)t * HIDC * HIDC,
        bl2 + (size_t)t * HIDC, out, C0N, t);
  }

  k_relu_ln<<<dim3((C0N + 7) / 8), blk, 0, stream>>>(out, g1, b1, C0N, 1.0f / 3.0f);
}
